// TransformerDecoderLayer_71279277244766
// MI455X (gfx1250) — compile-verified, matrix operand order checked by reading, not run on hardware
//
#include <hip/hip_runtime.h>
#include <cstdint>
#include <cstddef>

// ---------------------------------------------------------------------------
// Types for CDNA5 WMMA (wave32): bf16 A/B fragments (16 elems/lane), f32 C/D.
// ---------------------------------------------------------------------------
typedef __attribute__((ext_vector_type(16))) __bf16 v16bf;
typedef __attribute__((ext_vector_type(8)))  __bf16 v8bf;
typedef __attribute__((ext_vector_type(8)))  float  v8f;

#define DDIM  1024
#define NHEAD 16
#define HDIM  64
#define FDIM  4096
#define TSEQ  1024
#define BSZ   8
#define MROWS (BSZ * TSEQ)   // 8192

static __device__ __forceinline__ v16bf cat16(v8bf lo, v8bf hi) {
  return __builtin_shufflevector(lo, hi, 0,1,2,3,4,5,6,7,8,9,10,11,12,13,14,15);
}
static __device__ __forceinline__ v8f wmma_bf16(v16bf a, v16bf b, v8f c) {
  // D = A(16x32 bf16) x B(32x16 bf16) + C(16x16 f32)
  return __builtin_amdgcn_wmma_f32_16x16x32_bf16(false, a, false, b, (short)0, c,
                                                 false, false);
}

// ---------------------------------------------------------------------------
// CDNA5 async global->LDS copy (ASYNCcnt-tracked, no VGPR round trip).
// Generic LDS pointer: low 32 bits are the LDS byte offset (ISA 10.2 aperture).
// ---------------------------------------------------------------------------
static __device__ __forceinline__ void async_ld_b128(void* lds, const void* g) {
  uint32_t l = (uint32_t)(size_t)lds;
  uint64_t a = (uint64_t)(size_t)g;
  asm volatile("global_load_async_to_lds_b128 %0, %1, off" :: "v"(l), "v"(a)
               : "memory");
}
static __device__ __forceinline__ void wait_async0() {
  asm volatile("s_wait_asynccnt 0x0" ::: "memory");
}

// ---------------------------------------------------------------------------
// fp32 -> bf16 elementwise convert (weights / activations)
// ---------------------------------------------------------------------------
__global__ __launch_bounds__(256) void cvt_f32_bf16(const float* __restrict__ s,
                                                    __bf16* __restrict__ d, int n) {
  for (int i = blockIdx.x * 256 + threadIdx.x; i < n; i += gridDim.x * 256)
    d[i] = (__bf16)s[i];
}

// ---------------------------------------------------------------------------
// per-dim q scale: (log2e/sqrt(H)) * softplus(p)   (H = 64)
// ---------------------------------------------------------------------------
__global__ __launch_bounds__(64) void qscale_k(const float* __restrict__ p,
                                               float* __restrict__ qs) {
  int t = threadIdx.x;
  float v  = p[t];
  float sp = (v > 20.f) ? v : log1pf(__expf(v));
  qs[t] = 0.18033688f * sp;  // 1.442695041 / 8
}

// ---------------------------------------------------------------------------
// RMSNorm (row = one (b,t)) -> bf16
// ---------------------------------------------------------------------------
__global__ __launch_bounds__(256) void rmsnorm_bf16(const float* __restrict__ x,
                                                    const float* __restrict__ g,
                                                    __bf16* __restrict__ out) {
  const int row = blockIdx.x, t = threadIdx.x;
  const float* xr = x + (size_t)row * DDIM;
  float v[4], ss = 0.f;
#pragma unroll
  for (int i = 0; i < 4; i++) { v[i] = xr[t + 256 * i]; ss += v[i] * v[i]; }
#pragma unroll
  for (int off = 16; off; off >>= 1) ss += __shfl_xor(ss, off, 32);
  __shared__ float part[8];
  if ((t & 31) == 0) part[t >> 5] = ss;
  __syncthreads();
  if (t == 0) { float s = 0.f; for (int i = 0; i < 8; i++) s += part[i]; part[0] = s; }
  __syncthreads();
  const float inv = rsqrtf(part[0] * (1.f / DDIM) + 1e-6f);
  __bf16* orow = out + (size_t)row * DDIM;
#pragma unroll
  for (int i = 0; i < 4; i++) { int c = t + 256 * i; orow[c] = (__bf16)(v[i] * inv * g[c]); }
}

// ---------------------------------------------------------------------------
// LayerNorm -> bf16
// ---------------------------------------------------------------------------
__global__ __launch_bounds__(256) void layernorm_bf16(const float* __restrict__ x,
                                                      const float* __restrict__ g,
                                                      const float* __restrict__ b,
                                                      __bf16* __restrict__ out) {
  const int row = blockIdx.x, t = threadIdx.x;
  const float* xr = x + (size_t)row * DDIM;
  float v[4], s = 0.f, sq = 0.f;
#pragma unroll
  for (int i = 0; i < 4; i++) { v[i] = xr[t + 256 * i]; s += v[i]; sq += v[i] * v[i]; }
#pragma unroll
  for (int off = 16; off; off >>= 1) { s += __shfl_xor(s, off, 32); sq += __shfl_xor(sq, off, 32); }
  __shared__ float ps[8], pq[8];
  if ((t & 31) == 0) { ps[t >> 5] = s; pq[t >> 5] = sq; }
  __syncthreads();
  if (t == 0) {
    float a = 0.f, c = 0.f;
    for (int i = 0; i < 8; i++) { a += ps[i]; c += pq[i]; }
    ps[0] = a; pq[0] = c;
  }
  __syncthreads();
  const float mean = ps[0] * (1.f / DDIM);
  const float var  = pq[0] * (1.f / DDIM) - mean * mean;
  const float inv  = rsqrtf(var + 1e-5f);
  __bf16* orow = out + (size_t)row * DDIM;
#pragma unroll
  for (int i = 0; i < 4; i++) {
    int c = t + 256 * i;
    orow[c] = (__bf16)((v[i] - mean) * inv * g[c] + b[c]);
  }
}

// ---------------------------------------------------------------------------
// Tiled bf16 WMMA GEMM:  C[M,N] = A[M,K] * W + epilogue
//   WT=false : W is KxN row-major (transposed into LDS on stage)
//   WT=true  : W is NxK row-major (async-copied straight into LDS)
// Workgroup tile 128(M) x 128(N), 8 waves, each wave 32x64 (2x4 WMMA accs,
// 8 WMMAs per K-step). A tile always staged via async global->LDS b128.
// ---------------------------------------------------------------------------
template <bool WT, bool QS, bool RELU, bool RES, bool OUTF, bool OUTB>
__global__ __launch_bounds__(256) void gemm_bf16_k(
    const __bf16* __restrict__ A, const __bf16* __restrict__ W,
    const float* __restrict__ bias, const float* __restrict__ qsc,
    const float* __restrict__ res, float* __restrict__ outf,
    __bf16* __restrict__ outb, int M, int N, int K) {
  __shared__ __align__(16) __bf16 As[128 * 32];   // [m][k]
  __shared__ __align__(16) __bf16 Bs[128 * 32];   // [n][k]  (k-contiguous)
  const int tid  = threadIdx.x;
  const int w    = tid >> 5, lane = tid & 31;
  const int half = lane >> 4, ln = lane & 15;
  const int m0 = blockIdx.y * 128, n0 = blockIdx.x * 128;
  const int mo = (w & 3) * 32, no = (w >> 2) * 64;

  const v8f vz = {0.f, 0.f, 0.f, 0.f, 0.f, 0.f, 0.f, 0.f};
  v8f acc[2][4];
#pragma unroll
  for (int mi = 0; mi < 2; mi++)
#pragma unroll
    for (int ni = 0; ni < 4; ni++) acc[mi][ni] = vz;

  for (int k0 = 0; k0 < K; k0 += 32) {
    {  // stage A tile 128x32 via async DMA (2 x b128 per thread)
      int row = tid >> 1, kp = (tid & 1) * 16;
      const __bf16* ap = A + (size_t)(m0 + row) * K + k0 + kp;
      async_ld_b128(&As[row * 32 + kp],     ap);
      async_ld_b128(&As[row * 32 + kp + 8], ap + 8);
      if (k0 + 32 < K) __builtin_prefetch(ap + 32, 0, 0);
    }
    if (WT) {  // W already NxK: async straight copy (2 x b128 per thread)
      int nn = tid >> 1, kp = (tid & 1) * 16;
      const __bf16* wp = W + (size_t)(n0 + nn) * K + k0 + kp;
      async_ld_b128(&Bs[nn * 32 + kp],     wp);
      async_ld_b128(&Bs[nn * 32 + kp + 8], wp + 8);
      if (k0 + 32 < K) __builtin_prefetch(wp + 32, 0, 0);
    } else {   // W KxN: transpose on store (16 elems per thread)
      int kk = tid >> 3, nn = (tid & 7) * 16;
      const __bf16* wp = W + (size_t)(k0 + kk) * N + n0 + nn;
      v8bf t0 = *(const v8bf*)wp;
      v8bf t1 = *(const v8bf*)(wp + 8);
#pragma unroll
      for (int j = 0; j < 8; j++) {
        Bs[(nn + j) * 32 + kk]     = t0[j];
        Bs[(nn + 8 + j) * 32 + kk] = t1[j];
      }
      if (k0 + 32 < K) __builtin_prefetch(wp + (size_t)32 * N, 0, 0);
    }
    wait_async0();
    __syncthreads();

    v16bf af[2], bfm[4];
#pragma unroll
    for (int i = 0; i < 2; i++) {
      const __bf16* ap = &As[(mo + i * 16 + ln) * 32 + half * 8];
      af[i] = cat16(*(const v8bf*)ap, *(const v8bf*)(ap + 16));
    }
#pragma unroll
    for (int i = 0; i < 4; i++) {
      const __bf16* bp = &Bs[(no + i * 16 + ln) * 32 + half * 8];
      bfm[i] = cat16(*(const v8bf*)bp, *(const v8bf*)(bp + 16));
    }
#pragma unroll
    for (int mi = 0; mi < 2; mi++)
#pragma unroll
      for (int ni = 0; ni < 4; ni++)
        acc[mi][ni] = wmma_bf16(af[mi], bfm[ni], acc[mi][ni]);
    __syncthreads();
  }

  // epilogue: C layout -> lane ln is column, VGPR j + 8*half is row
#pragma unroll
  for (int mi = 0; mi < 2; mi++)
#pragma unroll
    for (int ni = 0; ni < 4; ni++) {
      const int gc = n0 + no + ni * 16 + ln;
      const float bv = bias[gc];
      const float qv = QS ? qsc[gc & (HDIM - 1)] : 1.f;
#pragma unroll
      for (int j = 0; j < 8; j++) {
        const int gr = m0 + mo + mi * 16 + 8 * half + j;
        float v = acc[mi][ni][j] + bv;
        if (QS)  v *= qv;
        if (RELU) v = fmaxf(v, 0.f);
        const size_t idx = (size_t)gr * N + gc;
        if (RES)  v += res[idx];
        if (OUTF) outf[idx] = v;
        if (OUTB) outb[idx] = (__bf16)v;
      }
    }
}

// ---------------------------------------------------------------------------
// Flash attention per (128-query tile, b, n). Q frags in registers; K staged
// row-major in LDS via async DMA; V transposed in LDS; online softmax; P via
// per-wave LDS round-trip. Layouts: Q/K/V/O all [b*T + t][n*64 + h] bf16.
// ---------------------------------------------------------------------------
__global__ __launch_bounds__(256) void attention_k(
    const __bf16* __restrict__ Q, const __bf16* __restrict__ K,
    const __bf16* __restrict__ V, const float* __restrict__ padding,
    __bf16* __restrict__ O) {
  const int qt = blockIdx.x;        // query tile (8 tiles of 128)
  const int bn = blockIdx.y;        // b*16 + n
  const int b = bn >> 4, n = bn & 15;
  const int tid = threadIdx.x, w = tid >> 5, lane = tid & 31;
  const int half = lane >> 4, ln = lane & 15;

  __shared__ __align__(16) __bf16 Ks[32 * 64];       // [s][h]
  __shared__ __align__(16) __bf16 Vt[64 * 32];       // [h][s]
  __shared__ __align__(16) __bf16 Pls[8][16 * 32];   // per-wave P tile
  __shared__ float Ps[32];

  const size_t base = (size_t)(b * TSEQ) * DDIM + n * HDIM;
  const int qbase = qt * 128 + w * 16;

  // Q fragments (two K-halves of H=64), resident for whole key loop
  v16bf qf[2];
  {
    const __bf16* qp = Q + base + (size_t)(qbase + ln) * DDIM;
#pragma unroll
    for (int hs = 0; hs < 2; hs++) {
      v8bf lo = *(const v8bf*)(qp + hs * 32 + half * 8);
      v8bf hi = *(const v8bf*)(qp + hs * 32 + 16 + half * 8);
      qf[hs] = cat16(lo, hi);
    }
  }
  float pq[8];
#pragma unroll
  for (int j = 0; j < 8; j++) pq[j] = padding[b * TSEQ + qbase + 8 * half + j];

  const v8f vz = {0.f, 0.f, 0.f, 0.f, 0.f, 0.f, 0.f, 0.f};
  v8f o[4]; o[0] = vz; o[1] = vz; o[2] = vz; o[3] = vz;
  float m_[8], l_[8];
#pragma unroll
  for (int j = 0; j < 8; j++) { m_[j] = -1e30f; l_[j] = 0.f; }

  const int smax = qt * 128 + 128;  // causal bound for this tile
  for (int s0 = 0; s0 < smax; s0 += 32) {
    {  // stage K block via async DMA; V block transposed manually
      int row = tid >> 3, ch = (tid & 7) * 8;
      async_ld_b128(&Ks[row * 64 + ch],
                    K + base + (size_t)(s0 + row) * DDIM + ch);
      v8bf vv = *(const v8bf*)(V + base + (size_t)(s0 + row) * DDIM + ch);
#pragma unroll
      for (int j = 0; j < 8; j++) Vt[(ch + j) * 32 + row] = vv[j];
      if (tid < 32) Ps[tid] = padding[b * TSEQ + s0 + tid];
      if (s0 + 32 < smax) {
        __builtin_prefetch(K + base + (size_t)(s0 + 32 + row) * DDIM + ch, 0, 0);
        __builtin_prefetch(V + base + (size_t)(s0 + 32 + row) * DDIM + ch, 0, 0);
      }
    }
    wait_async0();
    __syncthreads();

    // scores: two 16x16 tiles over 32 keys, K-dim = H = 64 (2 WMMAs each)
    v8f sa[2];
#pragma unroll
    for (int st = 0; st < 2; st++) {
      v8f accs = vz;
#pragma unroll
      for (int hs = 0; hs < 2; hs++) {
        const __bf16* kb = &Ks[(st * 16 + ln) * 64 + hs * 32 + half * 8];
        accs = wmma_bf16(qf[hs], cat16(*(const v8bf*)kb, *(const v8bf*)(kb + 16)), accs);
      }
      sa[st] = accs;
    }

    // mask + online softmax (row stats across 16-lane half-groups)
    float alpha[8];
#pragma unroll
    for (int j = 0; j < 8; j++) {
      const int qr = qbase + 8 * half + j;
      float v0 = sa[0][j], v1 = sa[1][j];
      const bool okq = (pq[j] != 0.f);
      v0 = (okq && (s0 + ln      <= qr) && (Ps[ln]      != 0.f)) ? v0 : -1e9f;
      v1 = (okq && (s0 + 16 + ln <= qr) && (Ps[16 + ln] != 0.f)) ? v1 : -1e9f;
      float mt = fmaxf(v0, v1);
#pragma unroll
      for (int off = 1; off < 16; off <<= 1) mt = fmaxf(mt, __shfl_xor(mt, off, 32));
      const float mn = fmaxf(m_[j], mt);
      alpha[j] = __expf(m_[j] - mn);
      float p0 = (v0 <= -1e8f) ? 0.f : __expf(v0 - mn);
      float p1 = (v1 <= -1e8f) ? 0.f : __expf(v1 - mn);
      float rs = p0 + p1;
#pragma unroll
      for (int off = 1; off < 16; off <<= 1) rs += __shfl_xor(rs, off, 32);
      l_[j] = l_[j] * alpha[j] + rs;
      m_[j] = mn;
      sa[0][j] = p0; sa[1][j] = p1;
    }
#pragma unroll
    for (int c = 0; c < 4; c++)
#pragma unroll
      for (int j = 0; j < 8; j++) o[c][j] *= alpha[j];

    // P: accumulator layout -> A-fragment layout via per-wave LDS tile
    __bf16* pl = &Pls[w][0];
#pragma unroll
    for (int j = 0; j < 8; j++) {
      pl[(8 * half + j) * 32 + ln]      = (__bf16)sa[0][j];
      pl[(8 * half + j) * 32 + 16 + ln] = (__bf16)sa[1][j];
    }
    v16bf pf;
    {
      const __bf16* pp = &Pls[w][ln * 32 + half * 8];
      pf = cat16(*(const v8bf*)pp, *(const v8bf*)(pp + 16));
    }
    // O += P(16x32) * V(32x64): four 16-wide column tiles
#pragma unroll
    for (int c = 0; c < 4; c++) {
      const __bf16* vb = &Vt[(c * 16 + ln) * 32 + half * 8];
      o[c] = wmma_bf16(pf, cat16(*(const v8bf*)vb, *(const v8bf*)(vb + 16)), o[c]);
    }
    __syncthreads();
  }

  // normalize and emit bf16 for the O-projection GEMM
#pragma unroll
  for (int j = 0; j < 8; j++) {
    const float inv = 1.f / fmaxf(l_[j], 1e-20f);
    const int qr = qbase + 8 * half + j;
    __bf16* op = O + base + (size_t)qr * DDIM;
#pragma unroll
    for (int c = 0; c < 4; c++) op[c * 16 + ln] = (__bf16)(o[c][j] * inv);
  }
}

// ---------------------------------------------------------------------------
// Host-side orchestration
// ---------------------------------------------------------------------------
extern "C" void kernel_launch(void* const* d_in, const int* in_sizes, int n_in,
                              void* d_out, int out_size, void* d_ws, size_t ws_size,
                              hipStream_t stream) {
  const float* x    = (const float*)d_in[0];
  const float* pad  = (const float*)d_in[1];
  const float* rmss = (const float*)d_in[2];
  const float* pds  = (const float*)d_in[3];
  const float* wq   = (const float*)d_in[4];
  const float* bq   = (const float*)d_in[5];
  const float* wk   = (const float*)d_in[6];
  const float* bk   = (const float*)d_in[7];
  const float* wv   = (const float*)d_in[8];
  const float* bv   = (const float*)d_in[9];
  const float* wo   = (const float*)d_in[10];
  const float* bo   = (const float*)d_in[11];
  const float* lng  = (const float*)d_in[12];
  const float* lnb  = (const float*)d_in[13];
  const float* w1   = (const float*)d_in[14];
  const float* b1   = (const float*)d_in[15];
  const float* w2   = (const float*)d_in[16];
  const float* b2   = (const float*)d_in[17];
  float* out = (float*)d_out;

  char* p = (char*)d_ws;
  auto take = [&](size_t bytes) -> char* {
    char* r = p;
    p += (bytes + 255) & ~(size_t)255;
    return r;
  };
  const size_t ME = (size_t)MROWS;            // 8192 rows
  __bf16* wqb = (__bf16*)take((size_t)DDIM * DDIM * 2);
  __bf16* wkb = (__bf16*)take((size_t)DDIM * DDIM * 2);
  __bf16* wvb = (__bf16*)take((size_t)DDIM * DDIM * 2);
  __bf16* wob = (__bf16*)take((size_t)DDIM * DDIM * 2);
  __bf16* w1b = (__bf16*)take((size_t)DDIM * FDIM * 2);
  __bf16* w2b = (__bf16*)take((size_t)FDIM * DDIM * 2);
  __bf16* xn  = (__bf16*)take(ME * DDIM * 2);
  __bf16* qb  = (__bf16*)take(ME * DDIM * 2);
  __bf16* kb  = (__bf16*)take(ME * DDIM * 2);
  __bf16* vb_ = (__bf16*)take(ME * DDIM * 2);
  __bf16* ao  = (__bf16*)take(ME * DDIM * 2);
  float*  at  = (float*)take(ME * DDIM * 4);  // attn + residual (fp32)
  __bf16* hn  = (__bf16*)take(ME * DDIM * 2);
  __bf16* h1  = (__bf16*)take(ME * FDIM * 2);
  float*  qsc = (float*)take(HDIM * 4);
  (void)in_sizes; (void)n_in; (void)out_size; (void)ws_size;

  // weight conversion fp32 -> bf16
  const int NW1 = DDIM * DDIM, NW2 = DDIM * FDIM;
  cvt_f32_bf16<<<(NW1 + 255) / 256, 256, 0, stream>>>(wq, wqb, NW1);
  cvt_f32_bf16<<<(NW1 + 255) / 256, 256, 0, stream>>>(wk, wkb, NW1);
  cvt_f32_bf16<<<(NW1 + 255) / 256, 256, 0, stream>>>(wv, wvb, NW1);
  cvt_f32_bf16<<<(NW1 + 255) / 256, 256, 0, stream>>>(wo, wob, NW1);
  cvt_f32_bf16<<<(NW2 + 255) / 256, 256, 0, stream>>>(w1, w1b, NW2);
  cvt_f32_bf16<<<(NW2 + 255) / 256, 256, 0, stream>>>(w2, w2b, NW2);
  qscale_k<<<1, 64, 0, stream>>>(pds, qsc);

  // RMSNorm -> bf16
  rmsnorm_bf16<<<MROWS, 256, 0, stream>>>(x, rmss, xn);

  // Q/K/V projections (Q gets per-dim scale folded in)
  dim3 g1(DDIM / 128, MROWS / 128);
  gemm_bf16_k<false, true,  false, false, false, true><<<g1, 256, 0, stream>>>(
      xn, wqb, bq, qsc, nullptr, nullptr, qb, MROWS, DDIM, DDIM);
  gemm_bf16_k<false, false, false, false, false, true><<<g1, 256, 0, stream>>>(
      xn, wkb, bk, nullptr, nullptr, nullptr, kb, MROWS, DDIM, DDIM);
  gemm_bf16_k<false, false, false, false, false, true><<<g1, 256, 0, stream>>>(
      xn, wvb, bv, nullptr, nullptr, nullptr, vb_, MROWS, DDIM, DDIM);

  // flash attention over all (b, n) heads
  attention_k<<<dim3(TSEQ / 128, BSZ * NHEAD), 256, 0, stream>>>(qb, kb, vb_, pad, ao);

  // O projection + residual (wo is [d][n*h] == N x K, pre-transposed)
  gemm_bf16_k<true, false, false, true, true, false><<<g1, 256, 0, stream>>>(
      ao, wob, bo, nullptr, x, at, nullptr, MROWS, DDIM, DDIM);

  // LayerNorm -> bf16
  layernorm_bf16<<<MROWS, 256, 0, stream>>>(at, lng, lnb, hn);

  // FFN
  dim3 g2(FDIM / 128, MROWS / 128);
  gemm_bf16_k<false, false, true, false, false, true><<<g2, 256, 0, stream>>>(
      hn, w1b, b1, nullptr, nullptr, nullptr, h1, MROWS, FDIM, DDIM);
  gemm_bf16_k<false, false, false, true, true, false><<<g1, 256, 0, stream>>>(
      h1, w2b, b2, nullptr, at, out, nullptr, MROWS, DDIM, FDIM);
}
